// CrossAttention_66073776881770
// MI455X (gfx1250) — matrix operand order repaired and checked by reading, not compiled, not run on hardware
//
#include <hip/hip_runtime.h>
#include <hip/hip_bf16.h>

typedef __attribute__((ext_vector_type(16))) __bf16 v16bf;
typedef __attribute__((ext_vector_type(8)))  float  v8f;
typedef __attribute__((ext_vector_type(4)))  int    v4i;

#define BB    2
#define NF    16
#define NP    196
#define DIMM  768
#define NH    12
#define DHD   64
#define TT    (NF * NP)      // 3136
#define MROWS (BB * TT)      // 6272
#define KV_LD (2 * DIMM)     // 1536
#define SCOLS 224            // 196 keys padded to 7*32
#define VSTR  72             // padded LDS row stride (bf16 elems) = 144B (16B-aligned)

// ---- gfx1250 async global->LDS path (guarded; falls back to sync copies) ----
#if defined(__has_builtin)
#  if __has_builtin(__builtin_amdgcn_global_load_async_to_lds_b128) && \
      __has_builtin(__builtin_amdgcn_s_wait_asynccnt)
#    define HAVE_ASYNC_LDS 1
#  endif
#endif
#ifndef HAVE_ASYNC_LDS
#  define HAVE_ASYNC_LDS 0
#endif

typedef __attribute__((address_space(1))) v4i* glb_v4i_ptr;
typedef __attribute__((address_space(3))) v4i* lds_v4i_ptr;

static __device__ inline glb_v4i_ptr as_glb_v4i(const void* p) {
  // AS1 addresses are identity with generic global addresses
  return (glb_v4i_ptr)(uintptr_t)p;
}
static __device__ inline lds_v4i_ptr as_lds_v4i(void* p) {
  // generic LDS address: low 32 bits are the LDS byte offset (ISA §10.2)
  return (lds_v4i_ptr)(uint32_t)(uintptr_t)p;
}

// ---------------------------------------------------------------------------
// fp32 -> bf16 elementwise convert
// ---------------------------------------------------------------------------
__global__ void cvt_bf16_kernel(const float* __restrict__ in,
                                __bf16* __restrict__ out, int n) {
  for (int i = blockIdx.x * blockDim.x + threadIdx.x; i < n;
       i += gridDim.x * blockDim.x)
    out[i] = (__bf16)in[i];
}

// ---------------------------------------------------------------------------
// W[K,N] fp32 -> Wt[N,K] bf16  (so WMMA B-fragments load contiguously in K)
// ---------------------------------------------------------------------------
__global__ void transpose_cvt_kernel(const float* __restrict__ W,
                                     __bf16* __restrict__ Wt, int K, int N) {
  for (int i = blockIdx.x * blockDim.x + threadIdx.x; i < K * N;
       i += gridDim.x * blockDim.x) {
    int n = i / K, k = i % K;
    Wt[i] = (__bf16)W[(size_t)k * N + n];
  }
}

// ---------------------------------------------------------------------------
// LayerNorm over last dim (768), biased var, eps=1e-5, output bf16
// ---------------------------------------------------------------------------
__global__ __launch_bounds__(256)
void ln_bf16_kernel(const float* __restrict__ x,
                    const float* __restrict__ gamma,
                    const float* __restrict__ beta,
                    __bf16* __restrict__ out) {
  int row = blockIdx.x, tid = threadIdx.x;
  const float* xr = x + (size_t)row * DIMM;
  float v0 = xr[tid], v1 = xr[tid + 256], v2 = xr[tid + 512];

  __shared__ float red[256];
  red[tid] = v0 + v1 + v2;
  __syncthreads();
  for (int o = 128; o > 0; o >>= 1) {
    if (tid < o) red[tid] += red[tid + o];
    __syncthreads();
  }
  float mu = red[0] * (1.0f / DIMM);
  __syncthreads();

  float d0 = v0 - mu, d1 = v1 - mu, d2 = v2 - mu;
  red[tid] = d0 * d0 + d1 * d1 + d2 * d2;
  __syncthreads();
  for (int o = 128; o > 0; o >>= 1) {
    if (tid < o) red[tid] += red[tid + o];
    __syncthreads();
  }
  float inv = rsqrtf(red[0] * (1.0f / DIMM) + 1e-5f);

  __bf16* orow = out + (size_t)row * DIMM;
  orow[tid]       = (__bf16)(d0 * inv * gamma[tid]       + beta[tid]);
  orow[tid + 256] = (__bf16)(d1 * inv * gamma[tid + 256] + beta[tid + 256]);
  orow[tid + 512] = (__bf16)(d2 * inv * gamma[tid + 512] + beta[tid + 512]);
}

// ---------------------------------------------------------------------------
// GEMM: C[M,N] = A[M,K] * Bt[N,K]^T, bf16 in / fp32 accumulate.
// Register-blocked: one wave computes a 64x64 C tile = 4x4 WMMA accumulators,
// 16 v_wmma_f32_16x16x32_bf16 per K-step of 32 (4x fragment reuse).
// M, N must be multiples of 64 (6272, 768, 1536 all are).
// ---------------------------------------------------------------------------
template <bool OUT_F32>
__global__ __launch_bounds__(256)
void gemm_bf16_wmma_kernel(const __bf16* __restrict__ A,
                           const __bf16* __restrict__ Bt,
                           int M, int N, int K,
                           __bf16* __restrict__ ob,
                           float* __restrict__ of,
                           const float* __restrict__ bias) {
  int lane = threadIdx.x & 31;
  int wave = blockIdx.x * (blockDim.x >> 5) + (threadIdx.x >> 5);
  int ntiles = N >> 6;
  int mt = wave / ntiles, nt = wave % ntiles;
  if (mt * 64 >= M) return;

  int l15 = lane & 15;
  int kb  = (lane >> 4) * 8;   // A-frag K base within 16-wide half
  int kh  = (lane >> 4) * 16;  // unused; B-frag also uses kb (same order as A)

  const __bf16* ar0 = A  + (size_t)(mt * 64 + l15) * K;
  const __bf16* br0 = Bt + (size_t)(nt * 64 + l15) * K;

  v8f acc[4][4] = {};
  for (int k0 = 0; k0 < K; k0 += 32) {
    __builtin_prefetch(ar0 + k0 + 128, 0, 1);
    __builtin_prefetch(br0 + k0 + 128, 0, 1);
    v16bf af[4];
#pragma unroll
    for (int i = 0; i < 4; ++i) {
      const __bf16* ar = ar0 + (size_t)(i * 16) * K;
#pragma unroll
      for (int e = 0; e < 8; ++e) {
        af[i][e]     = ar[k0 + kb + e];
        af[i][e + 8] = ar[k0 + 16 + kb + e];
      }
    }
#pragma unroll
    for (int j = 0; j < 4; ++j) {
      const __bf16* br = br0 + (size_t)(j * 16) * K;
      v16bf bf;
#pragma unroll
      for (int e = 0; e < 16; ++e) bf[e] = br[k0 + (e < 8 ? kb + e : 16 + kb + e - 8)];
#pragma unroll
      for (int i = 0; i < 4; ++i)
        acc[i][j] = __builtin_amdgcn_wmma_f32_16x16x32_bf16(
            false, af[i], false, bf, (short)0, acc[i][j], false, false);
    }
  }

#pragma unroll
  for (int i = 0; i < 4; ++i) {
#pragma unroll
    for (int j = 0; j < 4; ++j) {
      int ocol = nt * 64 + j * 16 + l15;
      float bv = 0.0f;
      if (OUT_F32) bv = bias ? bias[ocol] : 0.0f;
#pragma unroll
      for (int r8 = 0; r8 < 8; ++r8) {
        int r = mt * 64 + i * 16 + r8 + (lane >> 4) * 8;
        if (OUT_F32) of[(size_t)r * N + ocol] = acc[i][j][r8] + bv;
        else         ob[(size_t)r * N + ocol] = (__bf16)acc[i][j][r8];
      }
    }
  }
}

// ---------------------------------------------------------------------------
// Per-frame masked attention (mask == block-diagonal by frame).
// One wave per (b, h, frame, 16-query-row tile).
//  - V tile staged into LDS via global_load_async_to_lds_b128 (overlapped
//    with the Q*K^T WMMAs, waited with s_wait_asynccnt before P*V)
//  - S = QK^T: 13 col tiles x 2 WMMA;  P*V: 7 K-steps x 4 WMMA
// ---------------------------------------------------------------------------
__global__ __launch_bounds__(32)
void frame_attn_kernel(const __bf16* __restrict__ q,   // [B*T, 768] (h*64+d)
                       const __bf16* __restrict__ kv,  // [B*T, 1536] k|v
                       __bf16* __restrict__ o) {       // [B*T, 768]
  int lane = threadIdx.x;
  int id = blockIdx.x;
  int rt = id % 13; id /= 13;       // query row tile
  int f  = id % NF; id /= NF;       // frame
  int h  = id % NH;                 // head
  int b  = id / NH;                 // batch

  __shared__ float  S[16][SCOLS];       // 14.0 KB
  __shared__ __bf16 P[16][SCOLS];       //  7.0 KB
  __shared__ __bf16 Vlds[SCOLS][VSTR];  // 31.5 KB, rows 16B-aligned

  int l15 = lane & 15;
  int kb  = (lane >> 4) * 8;
  int kh  = (lane >> 4) * 16;
  size_t fbase = (size_t)(b * TT + f * NP);

  // ---- stage V tile (196x64 bf16, rows clamped to 224) into LDS ----
  // 8 x 16B chunks per row; issued early so the copy overlaps QK^T.
  for (int i = lane; i < SCOLS * 8; i += 32) {
    int r = i >> 3, c = i & 7;
    int kc = (r > NP - 1) ? (NP - 1) : r;
    const __bf16* g = kv + (fbase + kc) * KV_LD + DIMM + h * DHD + c * 8;
    __bf16* l = &Vlds[r][c * 8];
#if HAVE_ASYNC_LDS
    __builtin_amdgcn_global_load_async_to_lds_b128(as_glb_v4i(g),
                                                   as_lds_v4i(l), 0, 0);
#else
    *(uint4*)l = *(const uint4*)g;
#endif
  }

  // ---- Q A-fragments: row = query, K = head dim (64 -> 2 steps of 32) ----
  int qr = rt * 16 + l15; if (qr > NP - 1) qr = NP - 1;  // clamp pad rows
  const __bf16* qp = q + (fbase + qr) * DIMM + h * DHD;
  v16bf aq0, aq1;
#pragma unroll
  for (int e = 0; e < 8; ++e) {
    aq0[e]     = qp[kb + e];
    aq0[e + 8] = qp[16 + kb + e];
    aq1[e]     = qp[32 + kb + e];
    aq1[e + 8] = qp[48 + kb + e];
  }

  const float scale = 0.125f;  // 1/sqrt(64)

  // ---- S = (Q K^T) * scale, 13 column tiles of 16 keys ----
  for (int c = 0; c < 13; ++c) {
    int kc = c * 16 + l15; if (kc > NP - 1) kc = NP - 1;
    const __bf16* kp = kv + (fbase + kc) * KV_LD + h * DHD;  // key row
    v16bf bk0, bk1;
#pragma unroll
    for (int e = 0; e < 16; ++e) {
      bk0[e] = kp[(e < 8 ? kb + e : 16 + kb + e - 8)];
      bk1[e] = kp[32 + (e < 8 ? kb + e : 16 + kb + e - 8)];
    }
    v8f s = {};
    s = __builtin_amdgcn_wmma_f32_16x16x32_bf16(
        false, aq0, false, bk0, (short)0, s, false, false);
    s = __builtin_amdgcn_wmma_f32_16x16x32_bf16(
        false, aq1, false, bk1, (short)0, s, false, false);
#pragma unroll
    for (int j = 0; j < 8; ++j)
      S[j + (lane >> 4) * 8][c * 16 + l15] = s[j] * scale;
  }
  __syncthreads();

  // mask padded key columns (>=196)
  for (int i = lane; i < 16 * (SCOLS - NP); i += 32) {
    int r = i / (SCOLS - NP);
    int c = NP + i % (SCOLS - NP);
    S[r][c] = -1e30f;
  }
  __syncthreads();

  // ---- row softmax: 2 lanes per row (halves of 112 cols), shfl combine ----
  {
    int row = lane & 15, c0 = (lane >> 4) * (SCOLS / 2);
    float mx = -1e30f;
    for (int c = c0; c < c0 + SCOLS / 2; ++c) mx = fmaxf(mx, S[row][c]);
    mx = fmaxf(mx, __shfl_xor(mx, 16, 32));
    float sum = 0.0f;
    for (int c = c0; c < c0 + SCOLS / 2; ++c) {
      float p = __expf(S[row][c] - mx);
      S[row][c] = p;
      sum += p;
    }
    sum += __shfl_xor(sum, 16, 32);
    float inv = 1.0f / sum;
    for (int c = c0; c < c0 + SCOLS / 2; ++c)
      P[row][c] = (__bf16)(S[row][c] * inv);
  }
#if HAVE_ASYNC_LDS
  __builtin_amdgcn_s_wait_asynccnt(0);
#endif
  __syncthreads();

  // ---- O = P V : 7 K-steps of 32 keys x 4 output tiles (DH=64) ----
  v8f acc[4] = {};
  for (int ks = 0; ks < SCOLS; ks += 32) {
    v16bf ap;
#pragma unroll
    for (int e = 0; e < 8; ++e) {
      ap[e]     = P[l15][ks + kb + e];
      ap[e + 8] = P[l15][ks + 16 + kb + e];
    }
#pragma unroll
    for (int nt = 0; nt < 4; ++nt) {
      v16bf bv;
#pragma unroll
      for (int e = 0; e < 16; ++e)
        bv[e] = Vlds[ks + (e < 8 ? kb + e : 16 + kb + e - 8)][nt * 16 + l15];
      acc[nt] = __builtin_amdgcn_wmma_f32_16x16x32_bf16(
          false, ap, false, bv, (short)0, acc[nt], false, false);
    }
  }

#pragma unroll
  for (int nt = 0; nt < 4; ++nt)
#pragma unroll
    for (int j = 0; j < 8; ++j) {
      int r = rt * 16 + j + (lane >> 4) * 8;
      if (r < NP)
        o[(fbase + r) * DIMM + h * DHD + nt * 16 + l15] = (__bf16)acc[nt][j];
    }
}

// ---------------------------------------------------------------------------
extern "C" void kernel_launch(void* const* d_in, const int* in_sizes, int n_in,
                              void* d_out, int out_size, void* d_ws,
                              size_t ws_size, hipStream_t stream) {
  const float* x     = (const float*)d_in[0];
  const float* ctx   = (const float*)d_in[1];
  const float* Wq    = (const float*)d_in[2];
  const float* Wkv   = (const float*)d_in[3];
  const float* Wo    = (const float*)d_in[4];
  const float* bo    = (const float*)d_in[5];
  const float* gamma = (const float*)d_in[6];
  const float* beta  = (const float*)d_in[7];
  // d_in[8] = mask: block-diagonal by frame, applied structurally
  float* out = (float*)d_out;

  char* ws = (char*)d_ws;
  size_t off = 0;
  auto alloc = [&](size_t bytes) -> char* {
    char* p = ws + off;
    off += (bytes + 255) & ~(size_t)255;
    return p;
  };
  __bf16* Wqt  = (__bf16*)alloc((size_t)DIMM * DIMM * 2);   // [768,768]^T
  __bf16* Wkvt = (__bf16*)alloc((size_t)DIMM * KV_LD * 2);  // [1536,768]
  __bf16* Wot  = (__bf16*)alloc((size_t)DIMM * DIMM * 2);
  __bf16* xn   = (__bf16*)alloc((size_t)MROWS * DIMM * 2);
  __bf16* ctxb = (__bf16*)alloc((size_t)MROWS * DIMM * 2);
  __bf16* qb   = (__bf16*)alloc((size_t)MROWS * DIMM * 2);
  __bf16* kvb  = (__bf16*)alloc((size_t)MROWS * KV_LD * 2);
  __bf16* ab   = (__bf16*)alloc((size_t)MROWS * DIMM * 2);

  // 1) weight prep (transpose + bf16)
  transpose_cvt_kernel<<<2048, 256, 0, stream>>>(Wq,  Wqt,  DIMM, DIMM);
  transpose_cvt_kernel<<<2048, 256, 0, stream>>>(Wkv, Wkvt, DIMM, KV_LD);
  transpose_cvt_kernel<<<2048, 256, 0, stream>>>(Wo,  Wot,  DIMM, DIMM);

  // 2) LayerNorm(x) -> bf16 ; context -> bf16
  ln_bf16_kernel<<<MROWS, 256, 0, stream>>>(x, gamma, beta, xn);
  cvt_bf16_kernel<<<2048, 256, 0, stream>>>(ctx, ctxb, MROWS * DIMM);

  // 3) q = xn @ Wq   [6272,768] x [768,768]
  {
    int waves = (MROWS / 64) * (DIMM / 64);
    gemm_bf16_wmma_kernel<false><<<(waves + 7) / 8, 256, 0, stream>>>(
        xn, Wqt, MROWS, DIMM, DIMM, qb, nullptr, nullptr);
  }
  // 4) kv = ctx @ Wkv  [6272,768] x [768,1536]
  {
    int waves = (MROWS / 64) * (KV_LD / 64);
    gemm_bf16_wmma_kernel<false><<<(waves + 7) / 8, 256, 0, stream>>>(
        ctxb, Wkvt, MROWS, KV_LD, DIMM, kvb, nullptr, nullptr);
  }
  // 5) per-frame attention: B*H*NF*(13 row tiles) waves
  frame_attn_kernel<<<BB * NH * NF * 13, 32, 0, stream>>>(qb, kvb, ab);

  // 6) out = attn @ Wo + bo  (fp32 out)
  {
    int waves = (MROWS / 64) * (DIMM / 64);
    gemm_bf16_wmma_kernel<true><<<(waves + 7) / 8, 256, 0, stream>>>(
        ab, Wot, MROWS, DIMM, DIMM, nullptr, out, bo);
  }
}
